// Cross_At_50208167690362
// MI455X (gfx1250) — matrix operand order repaired and checked by reading, not compiled, not run on hardware
//
#include <hip/hip_runtime.h>
#include <hip/hip_bf16.h>
#include <math.h>

// ---------------------------------------------------------------------------
// Cross-attention (b=1, hw=4096, c=64, H=8, d=8), cosine-normalized, fused
// flash-attention on CDNA5 WMMA (v_wmma_f32_16x16x32_f16).
//
// Exact-math simplifications:
//  * scores = cos-sim in [-1,1] -> exp() cannot overflow -> no running max,
//    no correction factors, no per-tile reductions.
//  * V is N-padded 8->16 for WMMA anyway; dim 8 holds a column of ONES, so
//    the P*V accumulator's column 8 IS the softmax denominator (free).
//  * log2(e) folded into Q at pack time -> exp(s) == v_exp_f32(s) directly,
//    no per-score multiply.
// ---------------------------------------------------------------------------

#define HW 4096
#define C  64
#define NH 8
#define D  8
#define DP 16   // padded head dim (col 8 of V = ones, rest zero)
#define KT 64   // keys per main-loop iteration

#define LOG2E 1.4426950408889634f

typedef __attribute__((ext_vector_type(16))) _Float16 v16h;
typedef __attribute__((ext_vector_type(8)))  _Float16 v8h;
typedef __attribute__((ext_vector_type(8)))  float    v8f;

static __device__ inline v16h zero16h() {
  v16h z;
#pragma unroll
  for (int i = 0; i < 16; ++i) z[i] = (_Float16)0.f;
  return z;
}

// ---------------------------------------------------------------------------
// Kernel 1: q/k/v projections + per-head L2 normalize, pack to f16 workspace.
//   Qws, Kws : [NH][HW][DP] f16, cols 8..15 zeroed       (WMMA A / B tiles)
//   Vtws     : [NH][DP][HW] f16, row 8 = ones, 9..15 = 0 (WMMA B, transposed)
//   Q additionally scaled by log2(e) so the attention kernel can use exp2.
// ---------------------------------------------------------------------------
__global__ __launch_bounds__(256) void proj_kernel(
    const float* __restrict__ x, const float* __restrict__ y,
    const float* __restrict__ Wq, const float* __restrict__ bq,
    const float* __restrict__ Wkv, const float* __restrict__ bkv,
    _Float16* __restrict__ Qws, _Float16* __restrict__ Kws,
    _Float16* __restrict__ Vtws) {
  int gid   = blockIdx.x * 256 + threadIdx.x;   // 3*HW*NH threads
  int which = gid / (HW * NH);                  // 0=q, 1=k, 2=v
  int rem   = gid - which * (HW * NH);
  int h     = rem & (NH - 1);
  int row   = rem >> 3;

  const float* in = (which == 0) ? (x + row * C) : (y + row * C);
  const float* W  = (which == 0) ? Wq : Wkv;
  const float* bb = (which == 0) ? bq : bkv;
  int ldw   = (which == 0) ? C : 2 * C;
  int cbase = (which == 2) ? (C + h * D) : (h * D);

  float out[D];
#pragma unroll
  for (int j = 0; j < D; ++j) {
    int col = cbase + j;
    float a = bb[col];
    for (int i = 0; i < C; ++i) a = fmaf(in[i], W[i * ldw + col], a);
    out[j] = a;
  }

  if (which == 2) {
    // V: transposed [h][dim][row]; dim 8 = 1.0 (denominator column), 9..15 = 0.
#pragma unroll
    for (int j = 0; j < D; ++j) {
      Vtws[((size_t)h * DP + j) * HW + row]     = (_Float16)out[j];
      Vtws[((size_t)h * DP + D + j) * HW + row] = (_Float16)(j == 0 ? 1.f : 0.f);
    }
  } else {
    float ss = 0.f;
#pragma unroll
    for (int j = 0; j < D; ++j) ss += out[j] * out[j];
    float sc = 1.f / fmaxf(sqrtf(ss), 1e-12f);
    if (which == 0) sc *= LOG2E;   // fold exp->exp2 rescale into Q
    _Float16* dst = ((which == 0) ? Qws : Kws) + ((size_t)h * HW + row) * DP;
#pragma unroll
    for (int j = 0; j < D; ++j) {
      dst[j]     = (_Float16)(out[j] * sc);
      dst[D + j] = (_Float16)0.f;
    }
  }
}

// ---------------------------------------------------------------------------
// Kernel 2: fused attention.  grid = (HW/128, NH), block = 256 (8 waves).
// One wave = 16 query rows; key loop in tiles of 64:
//   4x WMMA  S = Q*K^T   (K-dim padded 8->32)
//   exp2(S) -> packed f16 -> per-wave LDS tile (C-layout -> A-layout)
//   2x WMMA  O += P*V    (V col 8 = ones accumulates the denominator)
// ---------------------------------------------------------------------------
__global__ __launch_bounds__(256) void attn_kernel(
    const _Float16* __restrict__ Q, const _Float16* __restrict__ K,
    const _Float16* __restrict__ Vt, float* __restrict__ AO) {
  __shared__ _Float16 Ptile[8][16][KT];   // per-wave P tile (16 KB total)

  int tid  = threadIdx.x;
  int wave = tid >> 5;
  int lane = tid & 31;
  int g    = lane >> 4;       // lane half-group (0 or 1)
  int ln   = lane & 15;
  int h    = blockIdx.y;
  int q0   = blockIdx.x * 128 + wave * 16;

  const _Float16* Qh = Q  + (size_t)h * HW * DP;
  const _Float16* Kh = K  + (size_t)h * HW * DP;
  const _Float16* Vh = Vt + (size_t)h * DP * HW;

  // ---- Q fragment (A matrix 16x32 f16): lanes 0-15 K=0..7 real / 16..23 zero,
  //      lanes 16-31 K=8..15 (zero-padded cols) / 24..31 zero.
  v16h qa;
  {
    v8h lo = *(const v8h*)(Qh + (size_t)(q0 + ln) * DP + 8 * g);
#pragma unroll
    for (int i = 0; i < 8; ++i) { qa[i] = lo[i]; qa[i + 8] = (_Float16)0.f; }
  }

  v8f acc = {};   // O accumulator; column 8 accumulates the softmax denominator

  for (int kb = 0; kb < HW; kb += KT) {
    // prefetch next iteration's K/V lines (speculative; safe past buffer end)
    __builtin_prefetch(Kh + (size_t)(kb + KT + ln) * DP);
    __builtin_prefetch(Vh + (size_t)ln * HW + kb + KT);

    // ---- 4 x 16-key S tiles:  S = Q * K^T  (S is already in log2 domain)
    v8f s[4];
#pragma unroll
    for (int t = 0; t < 4; ++t) {
      v16h kf;
      if (lane < 16) {
        kf = *(const v16h*)(Kh + (size_t)(kb + 16 * t + ln) * DP);
      } else {
        kf = zero16h();
      }
      v8f zc = {};
      s[t] = __builtin_amdgcn_wmma_f32_16x16x32_f16(false, qa, false, kf,
                                                    (short)0, zc, false, false);
    }

    // ---- P = exp2(S) (bounded), packed-convert to f16 pairs, store to LDS.
    //      C layout: vgpr r <-> row (r + 8*g), col = ln -> key = 16*t + ln.
#pragma unroll
    for (int t = 0; t < 4; ++t) {
#pragma unroll
      for (int r = 0; r < 8; r += 2) {
        auto hh = __builtin_amdgcn_cvt_pkrtz(__builtin_amdgcn_exp2f(s[t][r]),
                                             __builtin_amdgcn_exp2f(s[t][r + 1]));
        Ptile[wave][r + 8 * g][16 * t + ln]     = (_Float16)hh[0];
        Ptile[wave][r + 1 + 8 * g][16 * t + ln] = (_Float16)hh[1];
      }
    }

    // same-wave LDS store -> load ordering
    asm volatile("s_wait_dscnt 0" ::: "memory");

    // ---- two P*V WMMAs over key halves [kb..kb+31], [kb+32..kb+63]
#pragma unroll
    for (int u = 0; u < 2; ++u) {
      // P fragment (A 16x32): row m = ln; lanes 0-15: K=0..15,
      // lanes 16-31: K=16..31  (K relative to this 32-key half).
      const _Float16* pr = &Ptile[wave][ln][32 * u];
      v8h lo = *(const v8h*)(pr + 16 * g);
      v8h hi = *(const v8h*)(pr + 16 * g + 8);
      v16h pa;
#pragma unroll
      for (int i = 0; i < 8; ++i) { pa[i] = lo[i]; pa[i + 8] = hi[i]; }

      // V fragment (B 32x16): lane n = output dim; keys contiguous in Vt.
      v16h vb = *(const v16h*)(Vh + (size_t)ln * HW + kb + 32 * u + 16 * g);

      acc = __builtin_amdgcn_wmma_f32_16x16x32_f16(false, pa, false, vb,
                                                   (short)0, acc, false, false);
    }
  }

  // ---- epilogue: denominator = column 8 of each row (broadcast within the
  //      16-lane half-group), then scatter into merged [row][C] f32 buffer.
#pragma unroll
  for (int r = 0; r < 8; ++r) {
    float denom = __shfl(acc[r], 8 + 16 * g, 32);
    if (ln < D) {
      int row = q0 + r + 8 * g;
      AO[(size_t)row * C + h * D + ln] = acc[r] / denom;
    }
  }
}

// ---------------------------------------------------------------------------
// Kernel 3: output projection  out = AO @ We + be   (4096x64 @ 64x64)
// ---------------------------------------------------------------------------
__global__ __launch_bounds__(256) void outproj_kernel(
    const float* __restrict__ AO, const float* __restrict__ We,
    const float* __restrict__ be, float* __restrict__ out) {
  int gid = blockIdx.x * 256 + threadIdx.x;   // HW*C threads
  int row = gid >> 6;
  int col = gid & 63;
  const float* r = AO + (size_t)row * C;
  float a = be[col];
#pragma unroll 8
  for (int i = 0; i < C; ++i) a = fmaf(r[i], We[i * C + col], a);
  out[gid] = a;
}

// ---------------------------------------------------------------------------
extern "C" void kernel_launch(void* const* d_in, const int* in_sizes, int n_in,
                              void* d_out, int out_size, void* d_ws, size_t ws_size,
                              hipStream_t stream) {
  const float* x   = (const float*)d_in[0];
  const float* y   = (const float*)d_in[1];
  const float* Wq  = (const float*)d_in[2];
  const float* bq  = (const float*)d_in[3];
  const float* Wkv = (const float*)d_in[4];
  const float* bkv = (const float*)d_in[5];
  const float* We  = (const float*)d_in[6];
  const float* be  = (const float*)d_in[7];
  float* out = (float*)d_out;

  char* ws = (char*)d_ws;
  const size_t qk_bytes = (size_t)NH * HW * DP * sizeof(_Float16);  // 1 MB each
  _Float16* Qws  = (_Float16*)(ws);
  _Float16* Kws  = (_Float16*)(ws + qk_bytes);
  _Float16* Vtws = (_Float16*)(ws + 2 * qk_bytes);
  float*    AO   = (float*)   (ws + 3 * qk_bytes);                  // 1 MB

  // 1) projections + normalize + pack
  proj_kernel<<<(3 * HW * NH) / 256, 256, 0, stream>>>(
      x, y, Wq, bq, Wkv, bkv, Qws, Kws, Vtws);

  // 2) fused attention
  dim3 agrid(HW / 128, NH);
  attn_kernel<<<agrid, 256, 0, stream>>>(Qws, Kws, Vtws, AO);

  // 3) output projection
  outproj_kernel<<<(HW * C) / 256, 256, 0, stream>>>(AO, We, be, out);
}
